// ResidualQuantizer_721554506001
// MI455X (gfx1250) — matrix operand order repaired and checked by reading, not compiled, not run on hardware
//
#include <hip/hip_runtime.h>

// Residual VQ for MI455X (gfx1250, wave32, WMMA).
// - Distance GEMM on v_wmma_f32_16x16x32_bf16 with hi/lo error compensation
//   (3 bf16 WMMAs ~= f32 accuracy, ~2.7x fewer matrix instrs than f32 WMMA).
// - Codebook fragments staged per-workgroup into LDS with CDNA5 async
//   global->LDS copies (ASYNCcnt, double buffered) to cut L2 read demand 8x.

#define RVQ_B 131072
#define RVQ_D 256
#define RVQ_L 4
#define RVQ_K 1024

typedef __attribute__((ext_vector_type(16))) __bf16   v16bf;
typedef __attribute__((ext_vector_type(8)))  float    v8f;
typedef __attribute__((ext_vector_type(8)))  unsigned v8u;

union FragU { v8u u; v16bf b; };

__device__ __forceinline__ unsigned short f2bf(float f) {
  unsigned u = __builtin_bit_cast(unsigned, f);
  u += 0x7FFFu + ((u >> 16) & 1u);          // round-to-nearest-even
  return (unsigned short)(u >> 16);
}
__device__ __forceinline__ float bf2f(unsigned short s) {
  unsigned u = ((unsigned)s) << 16;
  return __builtin_bit_cast(float, u);
}

// Async copy of 16B global -> LDS (VFLAT GLOBAL_LOAD_ASYNC_TO_LDS_B128).
// Tracked by ASYNCcnt; lds/global both get INST_OFFSET added.
__device__ __forceinline__ void async_cp_b128(unsigned lds_addr, const void* gptr) {
  asm volatile("global_load_async_to_lds_b128 %0, %1, off"
               :: "v"(lds_addr), "v"((unsigned long long)(size_t)gptr)
               : "memory");
}
__device__ __forceinline__ void async_cp_b128_off16(unsigned lds_addr, const void* gptr) {
  asm volatile("global_load_async_to_lds_b128 %0, %1, off offset:16"
               :: "v"(lds_addr), "v"((unsigned long long)(size_t)gptr)
               : "memory");
}

// ---------------------------------------------------------------------------
// Prep 1: per-code squared norms + zero the loss accumulators (every call).
// ---------------------------------------------------------------------------
__global__ void rvq_prep_cnorm(const float* __restrict__ cb,
                               float* __restrict__ cnorm,
                               float* __restrict__ lossacc) {
  int t = blockIdx.x * blockDim.x + threadIdx.x;   // L*K = 4096 rows
  if (t < RVQ_L * RVQ_K) {
    const float4* row = (const float4*)(cb + (size_t)t * RVQ_D);
    float s = 0.f;
    for (int q = 0; q < RVQ_D / 4; ++q) {
      float4 v = row[q];
      s += v.x * v.x + v.y * v.y + v.z * v.z + v.w * v.w;
    }
    cnorm[t] = s;
  }
  if (blockIdx.x == 0 && threadIdx.x < RVQ_L) lossacc[threadIdx.x] = 0.f;
}

// ---------------------------------------------------------------------------
// Prep 2: codebook -> bf16 hi/lo B-fragments, pre-swizzled for the WMMA
// 32x16 bf16 B operand in contiguous k order, same as A (lane = 16*half + n, dword j holds K = 16*half + 2j).
// Stored lane-major (32B per lane): tile-contiguous 8KB blocks for async copy.
// ---------------------------------------------------------------------------
__global__ void rvq_prep_frags(const float* __restrict__ cb,
                               unsigned* __restrict__ fragHi,
                               unsigned* __restrict__ fragLo) {
  int tid = blockIdx.x * blockDim.x + threadIdx.x;  // L*K*128 = 524288
  int dp = tid & 127;
  int k  = (tid >> 7) & 1023;
  int l  = tid >> 17;
  int d0 = dp * 2;
  const float* src = cb + ((size_t)l * RVQ_K + k) * RVQ_D + d0;
  float c0 = src[0], c1 = src[1];
  unsigned short h0 = f2bf(c0), h1 = f2bf(c1);
  unsigned short g0 = f2bf(c0 - bf2f(h0)), g1 = f2bf(c1 - bf2f(h1));

  int kt = k >> 4, n = k & 15;
  int dc = d0 >> 5, dk = d0 & 31;
  int half = dk >> 4;
  int j    = (dk & 15) >> 1;
  int lane = half * 16 + n;
  size_t off = ((size_t)((l * 64 + kt) * 8 + dc)) * 256 + (size_t)lane * 8 + j;
  fragHi[off] = (unsigned)h0 | ((unsigned)h1 << 16);
  fragLo[off] = (unsigned)g0 | ((unsigned)g1 << 16);
}

// ---------------------------------------------------------------------------
// One RVQ level: fused score-GEMM (WMMA bf16 x3 compensated) + argmin +
// gather + residual update + loss partial sums.
// - Each wave owns 16 rows; A fragments (hi/lo, 8 d-chunks) live in VGPRs.
// - B fragments: one async global->LDS stage per WG per K-tile (16KB),
//   double buffered, consumed by all 8 waves via ds_load_b128.
// ---------------------------------------------------------------------------
__global__ void __launch_bounds__(256, 1)
rvq_level(const float* __restrict__ R,          // residual in [B, 256]
          const float* __restrict__ cbf32,      // f32 codebooks [L,K,256]
          const unsigned* __restrict__ fragHi,
          const unsigned* __restrict__ fragLo,
          const float* __restrict__ cnorm,      // [L*K]
          int level,
          int* __restrict__ idx_out,            // [B, L]
          float* __restrict__ qstack,           // [B, L, 256]
          float* __restrict__ resout,           // [B, 256] (residual_states[l+1])
          float* __restrict__ lossacc) {
  __shared__ __align__(16) unsigned char smem[2 * 16384];  // double buffer

  const int lane  = threadIdx.x & 31;
  const int nlane = lane & 15;
  const int half  = lane >> 4;
  const int wid   = blockIdx.x * (blockDim.x >> 5) + (threadIdx.x >> 5);
  const int rowbase = wid * 16;

  const unsigned* lvlH = fragHi + (size_t)(level * 64) * 2048;
  const unsigned* lvlL = fragLo + (size_t)(level * 64) * 2048;

  // Issue the 16KB tile copy for K-tile kt into LDS buffer par (4 x b128/thread).
  auto copy_tile = [&](int kt, int par) {
    const int t = threadIdx.x;
    const unsigned* gh = lvlH + (size_t)kt * 2048 + (size_t)t * 8;  // 32B/thread
    const unsigned* gl = lvlL + (size_t)kt * 2048 + (size_t)t * 8;
    unsigned dh = (unsigned)(size_t)(smem + (size_t)par * 16384 + (size_t)t * 32);
    unsigned dl = dh + 8192;
    async_cp_b128(dh, gh);
    async_cp_b128_off16(dh, gh);
    async_cp_b128(dl, gl);
    async_cp_b128_off16(dl, gl);
  };

  // --- Build A fragments (16x32 bf16 A, contiguous k order matching B: lane m = nlane, half picks
  //     K = 16*half..16*half+15; dword j holds K pair kb, kb+1). ---
  v16bf aHi[8], aLo[8];
  {
    const float* rrow = R + (size_t)(rowbase + nlane) * RVQ_D;
#pragma unroll
    for (int dc = 0; dc < 8; ++dc) {
      FragU uh, ul;
#pragma unroll
      for (int j = 0; j < 8; ++j) {
        int kb = 16 * half + 2 * j;
        float c0 = rrow[dc * 32 + kb];
        float c1 = rrow[dc * 32 + kb + 1];
        unsigned short h0 = f2bf(c0), h1 = f2bf(c1);
        unsigned short g0 = f2bf(c0 - bf2f(h0)), g1 = f2bf(c1 - bf2f(h1));
        uh.u[j] = (unsigned)h0 | ((unsigned)h1 << 16);
        ul.u[j] = (unsigned)g0 | ((unsigned)g1 << 16);
      }
      aHi[dc] = uh.b;
      aLo[dc] = ul.b;
    }
  }

  // --- K loop: 64 tiles of 16 codes; running argmin per accumulator row. ---
  float best[8];
  int   bidx[8];
#pragma unroll
  for (int g = 0; g < 8; ++g) { best[g] = 3.4e38f; bidx[g] = 0; }

  copy_tile(0, 0);                          // prime the pipeline

  for (int kt = 0; kt < 64; ++kt) {
    if (kt + 1 < 64) {
      copy_tile(kt + 1, (kt + 1) & 1);      // prefetch next tile
      asm volatile("s_wait_asynccnt 0x4" ::: "memory");  // current tile landed
    } else {
      asm volatile("s_wait_asynccnt 0x0" ::: "memory");
    }
    __syncthreads();                        // all waves' copies visible

    float cn = cnorm[level * RVQ_K + kt * 16 + nlane];
    const unsigned char* bufp = smem + (size_t)(kt & 1) * 16384;

    // Two independent WMMA chains: acc0 = hi*hi, acc1 = hi*lo + lo*hi.
    v8f acc0 = {0.f, 0.f, 0.f, 0.f, 0.f, 0.f, 0.f, 0.f};
    v8f acc1 = {0.f, 0.f, 0.f, 0.f, 0.f, 0.f, 0.f, 0.f};
#pragma unroll
    for (int dc = 0; dc < 8; ++dc) {
      FragU bh, bl;
      bh.u = *((const v8u*)(bufp + dc * 1024 + lane * 32));
      bl.u = *((const v8u*)(bufp + 8192 + dc * 1024 + lane * 32));
      acc0 = __builtin_amdgcn_wmma_f32_16x16x32_bf16(false, aHi[dc], false, bh.b,
                                                     (short)0, acc0, false, false);
      acc1 = __builtin_amdgcn_wmma_f32_16x16x32_bf16(false, aHi[dc], false, bl.b,
                                                     (short)0, acc1, false, false);
      acc1 = __builtin_amdgcn_wmma_f32_16x16x32_bf16(false, aLo[dc], false, bh.b,
                                                     (short)0, acc1, false, false);
    }

    int kc = kt * 16 + nlane;
#pragma unroll
    for (int g = 0; g < 8; ++g) {
      float dist = cn - 2.0f * (acc0[g] + acc1[g]);  // ||r||^2 per-row: dropped
      if (dist < best[g]) { best[g] = dist; bidx[g] = kc; }
    }
    __syncthreads();                        // reads done before buffer reuse
  }

  // --- Argmin across the 16 lanes of each half (row m = g + 8*half). ---
#pragma unroll
  for (int g = 0; g < 8; ++g) {
    float v = best[g];
    int   i = bidx[g];
#pragma unroll
    for (int off = 1; off <= 8; off <<= 1) {
      float vo = __shfl_xor(v, off, 32);
      int   io = __shfl_xor(i, off, 32);
      if (vo < v || (vo == v && io < i)) { v = vo; i = io; }
    }
    best[g] = v;
    bidx[g] = i;
  }
  if (nlane == 0) {
#pragma unroll
    for (int g = 0; g < 8; ++g)
      idx_out[(size_t)(rowbase + g + 8 * half) * RVQ_L + level] = bidx[g];
  }

  // --- Gather + residual + loss (all 32 lanes cooperate per row). ---
  float lsum = 0.f;
  auto dorow = [&](int b, int kq) {
    const float4* qp = (const float4*)(cbf32 + ((size_t)level * RVQ_K + kq) * RVQ_D);
    const float4* rp = (const float4*)(R + (size_t)b * RVQ_D);
    float4* qs = (float4*)(qstack + ((size_t)b * RVQ_L + level) * RVQ_D);
    float4* ro = (float4*)(resout + (size_t)b * RVQ_D);
#pragma unroll
    for (int tt = 0; tt < 2; ++tt) {
      int t = lane + tt * 32;           // 64 float4 per row
      float4 q = qp[t], r = rp[t];
      float4 nr;
      nr.x = r.x - q.x; nr.y = r.y - q.y; nr.z = r.z - q.z; nr.w = r.w - q.w;
      qs[t] = q;
      ro[t] = nr;
      lsum += nr.x * nr.x + nr.y * nr.y + nr.z * nr.z + nr.w * nr.w;
    }
  };
#pragma unroll
  for (int g = 0; g < 8; ++g) {
    int k0 = __shfl(bidx[g], 0, 32);    // half-0 result -> rows g
    int k1 = __shfl(bidx[g], 16, 32);   // half-1 result -> rows g+8
    dorow(rowbase + g, k0);
    dorow(rowbase + g + 8, k1);
  }
#pragma unroll
  for (int off = 1; off <= 16; off <<= 1) lsum += __shfl_xor(lsum, off, 32);
  if (lane == 0) atomicAdd(lossacc + level, lsum);
}

// ---------------------------------------------------------------------------
// Epilogue: quantized_sum (= sum of levels), residual_states[0] = ze, scalars.
// ---------------------------------------------------------------------------
__global__ void rvq_epilogue(const float4* __restrict__ ze4,
                             const float4* __restrict__ stack4,
                             float4* __restrict__ qsum4,
                             float4* __restrict__ qsumq4,
                             float4* __restrict__ res0_4,
                             const float* __restrict__ lossacc,
                             float* __restrict__ scal,
                             size_t n4) {
  size_t i = (size_t)blockIdx.x * blockDim.x + threadIdx.x;
  if (i < n4) {
    size_t b = i >> 6;                   // 64 float4 per row
    int    t = (int)(i & 63);
    float4 s = {0.f, 0.f, 0.f, 0.f};
#pragma unroll
    for (int l = 0; l < RVQ_L; ++l) {
      float4 q = stack4[(b * RVQ_L + l) * 64 + t];
      s.x += q.x; s.y += q.y; s.z += q.z; s.w += q.w;
    }
    qsum4[i]  = s;
    qsumq4[i] = s;
    res0_4[i] = ze4[i];
  }
  if (blockIdx.x == 0 && threadIdx.x == 0) {
    float c = (lossacc[0] + lossacc[1] + lossacc[2] + lossacc[3]) *
              (1.0f / ((float)RVQ_L * (float)RVQ_B * (float)RVQ_D));
    scal[0] = c;            // commitment_loss
    scal[1] = c;            // codebook_loss (numerically identical)
    scal[2] = 1.25f * c;    // 0.25*commit + codebook
  }
}

// ---------------------------------------------------------------------------
extern "C" void kernel_launch(void* const* d_in, const int* in_sizes, int n_in,
                              void* d_out, int out_size, void* d_ws, size_t ws_size,
                              hipStream_t stream) {
  const float* ze = (const float*)d_in[0];     // [B, D] f32
  const float* cb = (const float*)d_in[1];     // [L, K, D] f32

  float* out     = (float*)d_out;
  int*   o_idx   = (int*)out;                                    // [B, L]
  float* o_stack = out + (size_t)RVQ_B * RVQ_L;                  // [B, L, D]
  float* o_qsum  = o_stack + (size_t)RVQ_B * RVQ_L * RVQ_D;      // [B, D]
  float* o_qsumq = o_qsum + (size_t)RVQ_B * RVQ_D;               // [B, D]
  float* o_res   = o_qsumq + (size_t)RVQ_B * RVQ_D;              // [L+1, B, D]
  float* o_scal  = o_res + (size_t)(RVQ_L + 1) * RVQ_B * RVQ_D;  // 3 scalars

  float* ws      = (float*)d_ws;
  float* cnorm   = ws;                         // 4096 floats
  float* lossacc = ws + 4096;                  // 4 floats
  unsigned* fragHi = (unsigned*)(ws + 4352);   // L*64*8*256 dwords
  unsigned* fragLo = fragHi + (size_t)RVQ_L * 64 * 8 * 256;

  rvq_prep_cnorm<<<(RVQ_L * RVQ_K + 255) / 256, 256, 0, stream>>>(cb, cnorm, lossacc);
  rvq_prep_frags<<<(RVQ_L * RVQ_K * 128) / 256, 256, 0, stream>>>(cb, fragHi, fragLo);

  for (int l = 0; l < RVQ_L; ++l) {
    const float* Rin = (l == 0) ? ze : (o_res + (size_t)l * RVQ_B * RVQ_D);
    float* resout = o_res + (size_t)(l + 1) * RVQ_B * RVQ_D;
    // B/16 waves, 8 waves (256 threads) per workgroup -> B/128 blocks
    rvq_level<<<RVQ_B / 128, 256, 0, stream>>>(Rin, cb, fragHi, fragLo, cnorm, l,
                                               o_idx, o_stack, resout, lossacc);
  }

  size_t n4 = (size_t)RVQ_B * RVQ_D / 4;
  rvq_epilogue<<<(unsigned)((n4 + 255) / 256), 256, 0, stream>>>(
      (const float4*)ze, (const float4*)o_stack, (float4*)o_qsum,
      (float4*)o_qsumq, (float4*)o_res, lossacc, o_scal, n4);
}
